// MultiHeadAttentionBlock_1829656068307
// MI455X (gfx1250) — compile-verified, matrix operand order checked by reading, not run on hardware
//
#include <hip/hip_runtime.h>
#include <hip/hip_bf16.h>

// ---------------------------------------------------------------------------
// MultiHeadAttentionBlock on gfx1250 (MI455X): WMMA f16 (f32 accum) everywhere.
// b=4, c=128, seq=2048, heads=4, dk=32.
// ---------------------------------------------------------------------------

typedef __attribute__((ext_vector_type(16))) _Float16     v16h;
typedef __attribute__((ext_vector_type(8)))  float        v8f;
typedef __attribute__((ext_vector_type(4)))  unsigned int u32x4;

constexpr int BATCH = 4;
constexpr int CCH   = 128;   // channels c
constexpr int SEQ   = 2048;  // D*H*W
constexpr int NH    = 4;     // heads
constexpr int DK    = 32;    // head dim
constexpr int QKW   = 2 * DK * NH;  // 256: q,k packed per token (64 per head)
constexpr int PSTR  = 40;    // LDS P-tile row stride in halves (80B, 16B-mult)

// q is pre-scaled by (1/sqrt(dk)) * log2(e) so attention exponentials are
// plain exp2 (v_exp_f32 without the extra v_mul): log2(e)/sqrt(32).
#define QSCALE 0.2550348626f

// Load a 16-bit A/B fragment for v_wmma_*_16x16x32 from a contiguous 32-half
// row/column.  Per ISA 7.12.2: lane (hi = lane>=16) holds K indices
// {hi*8..hi*8+7} in elements 0..7 and {16+hi*8 .. 16+hi*8+7} in elements 8..15.
__device__ __forceinline__ v16h load_half_frag(const _Float16* p, int hi) {
  union { v16h v; u32x4 q[2]; } u;
  u.q[0] = *(const u32x4*)(p + hi * 8);
  u.q[1] = *(const u32x4*)(p + 16 + hi * 8);
  return u.v;
}

// Same fragment, but converting from a contiguous f32 row on the fly.
__device__ __forceinline__ v16h load_f32_frag(const float* p, int hi) {
  v16h f;
#pragma unroll
  for (int j = 0; j < 8; ++j) {
    f[j]     = (_Float16)p[hi * 8 + j];
    f[8 + j] = (_Float16)p[16 + hi * 8 + j];
  }
  return f;
}

// ---------------------------------------------------------------------------
// Kernel 1: fused QKV projection.
//   qkv[t, n] = sum_k x[b, k, t] * Wp[n, k] + bp[n]     (n in 0..383)
// D-tile = A(16 tokens x 32 ch) x B(32 ch x 16 outch), 4 K-steps.
// q (pre-scaled by QSCALE) / k -> qkH (b, seq, 256) half, token-major
// v -> vT (b, h, dk, seq) half (key-major rows for P*V B-fragments)
// ---------------------------------------------------------------------------
__global__ __launch_bounds__(32) void qkv_proj_kernel(
    const float* __restrict__ x, const float* __restrict__ Wp,
    const float* __restrict__ bp, _Float16* __restrict__ qkH,
    _Float16* __restrict__ vT) {
  const int lane = threadIdx.x;
  const int hi   = lane >> 4;
  const int col  = lane & 15;
  const int i0   = blockIdx.x * 16;  // token tile
  const int n0   = blockIdx.y * 16;  // qkv channel tile (0..383)
  const int bb   = blockIdx.z;

  const float* wrow  = Wp + (size_t)(n0 + col) * CCH;  // B column n=col
  const float* xbase = x + (size_t)bb * CCH * SEQ;
  const int tok = i0 + col;                            // A row m=col

  v8f acc = {};
#pragma unroll
  for (int kc = 0; kc < CCH; kc += 32) {
    v16h a;
#pragma unroll
    for (int j = 0; j < 8; ++j) {
      const int k0 = kc + hi * 8 + j;
      const int k1 = kc + 16 + hi * 8 + j;
      a[j]     = (_Float16)xbase[(size_t)k0 * SEQ + tok];
      a[8 + j] = (_Float16)xbase[(size_t)k1 * SEQ + tok];
    }
    v16h b = load_f32_frag(wrow + kc, hi);
    acc = __builtin_amdgcn_wmma_f32_16x16x32_f16(false, a, false, b,
                                                 (short)0, acc, false, false);
  }

  const int   n    = n0 + col;
  const float bias = bp[n];
  const int   head = n / 96;
  const int   rem  = n % 96;
  const int   part = rem >> 5;  // 0=q, 1=k, 2=v
  const int   d    = rem & 31;
  const float post = (part == 0) ? QSCALE : 1.0f;  // fold softmax scale into q
#pragma unroll
  for (int r = 0; r < 8; ++r) {
    const int t = i0 + r + hi * 8;            // D element (m=r+hi*8, n=col)
    const _Float16 hv = (_Float16)((acc[r] + bias) * post);
    if (part < 2) {
      qkH[((size_t)(bb * SEQ + t)) * QKW + head * 64 + part * 32 + d] = hv;
    } else {
      vT[(((size_t)(bb * NH + head)) * DK + d) * SEQ + t] = hv;
    }
  }
}

// ---------------------------------------------------------------------------
// Online-softmax update (log2 domain) for one 16-query row set against a
// 32-key chunk.  s0/s1 are pre-scaled S tiles (C-frag layout).  Writes
// exp2-probabilities (P) into the LDS tile; row sums are accumulated later
// via a WMMA against an all-ones fragment.
// ---------------------------------------------------------------------------
__device__ __forceinline__ void softmax_update(
    const v8f& s0, const v8f& s1, float* mrow, float* lrow, v8f& acc0,
    v8f& acc1, _Float16* prow, int hi, int col) {
  float mnew[8];
#pragma unroll
  for (int r = 0; r < 8; ++r) mnew[r] = fmaxf(s0[r], s1[r]);
  // Row max across the 16 lanes holding this row's columns.
#pragma unroll
  for (int off = 1; off < 16; off <<= 1)
#pragma unroll
    for (int r = 0; r < 8; ++r)
      mnew[r] = fmaxf(mnew[r], __shfl_xor(mnew[r], off, 32));

#pragma unroll
  for (int r = 0; r < 8; ++r) {
    mnew[r] = fmaxf(mnew[r], mrow[r]);
    const float alpha = exp2f(mrow[r] - mnew[r]);
    const float p0 = exp2f(s0[r] - mnew[r]);
    const float p1 = exp2f(s1[r] - mnew[r]);
    lrow[r] *= alpha;
    acc0[r] *= alpha;
    acc1[r] *= alpha;
    mrow[r] = mnew[r];
    const int m = r + hi * 8;
    prow[m * PSTR + col]      = (_Float16)p0;
    prow[m * PSTR + 16 + col] = (_Float16)p1;
  }
}

// ---------------------------------------------------------------------------
// Kernel 2: flash attention per (b, head, 32-query tile).
// Two Q fragments per wave; each 32-key chunk does 10 WMMAs (4 S, 4 P*V,
// 2 row-sum) against a single set of K/V fragment loads.  P is re-laid out
// C-frag -> A-frag through LDS.  Next chunk's K/V rows are prefetched.
// O -> oH (b, seq, 128) half.
// ---------------------------------------------------------------------------
__global__ __launch_bounds__(32) void attn_kernel(
    const _Float16* __restrict__ qkH, const _Float16* __restrict__ vT,
    _Float16* __restrict__ oH) {
  __shared__ _Float16 Plds[32 * PSTR];  // 32x32 P tile, padded rows

  const int lane = threadIdx.x;
  const int hi   = lane >> 4;
  const int col  = lane & 15;
  const int i0   = blockIdx.x * 32;
  const int head = blockIdx.y;
  const int bb   = blockIdx.z;

  // Q A-fragments: row m=col -> tokens i0+col and i0+16+col.
  const _Float16* qrow =
      qkH + ((size_t)(bb * SEQ + i0 + col)) * QKW + head * 64;
  const v16h aq0 = load_half_frag(qrow, hi);
  const v16h aq1 = load_half_frag(qrow + (size_t)16 * QKW, hi);

  // V^T rows for the two dk halves: column n=col -> d = col / 16+col.
  const _Float16* vrow0 = vT + (((size_t)(bb * NH + head)) * DK + col) * SEQ;
  const _Float16* vrow1 = vrow0 + (size_t)16 * SEQ;

  // All-ones B fragment: row sum of P via WMMA (P x ones -> sums in C layout).
  v16h bones;
#pragma unroll
  for (int j = 0; j < 16; ++j) bones[j] = (_Float16)1.0f;

  v8f acc00 = {}, acc01 = {}, acc10 = {}, acc11 = {};
  float mrow0[8], lrow0[8], mrow1[8], lrow1[8];
#pragma unroll
  for (int r = 0; r < 8; ++r) {
    mrow0[r] = -1e30f; lrow0[r] = 0.f;
    mrow1[r] = -1e30f; lrow1[r] = 0.f;
  }

  for (int j0 = 0; j0 < SEQ; j0 += 32) {
    // K B-fragments: column n -> token j0+n, contiguous 32 halves (k part).
    const _Float16* kbase =
        qkH + ((size_t)(bb * SEQ + j0 + col)) * QKW + head * 64 + 32;
    const v16h bk0 = load_half_frag(kbase, hi);
    const v16h bk1 = load_half_frag(kbase + (size_t)16 * QKW, hi);

    if (j0 + 32 < SEQ) {  // prefetch next chunk (global_prefetch_b8)
      __builtin_prefetch(kbase + (size_t)32 * QKW, 0, 0);
      __builtin_prefetch(vrow0 + j0 + 32, 0, 0);
      __builtin_prefetch(vrow1 + j0 + 32, 0, 0);
    }

    v8f z = {};
    v8f s00 = __builtin_amdgcn_wmma_f32_16x16x32_f16(false, aq0, false, bk0,
                                                     (short)0, z, false, false);
    v8f s01 = __builtin_amdgcn_wmma_f32_16x16x32_f16(false, aq0, false, bk1,
                                                     (short)0, z, false, false);
    v8f s10 = __builtin_amdgcn_wmma_f32_16x16x32_f16(false, aq1, false, bk0,
                                                     (short)0, z, false, false);
    v8f s11 = __builtin_amdgcn_wmma_f32_16x16x32_f16(false, aq1, false, bk1,
                                                     (short)0, z, false, false);

    softmax_update(s00, s01, mrow0, lrow0, acc00, acc01, Plds, hi, col);
    softmax_update(s10, s11, mrow1, lrow1, acc10, acc11, Plds + 16 * PSTR,
                   hi, col);
    __syncthreads();

    const v16h ap0 = load_half_frag(&Plds[col * PSTR], hi);
    const v16h ap1 = load_half_frag(&Plds[(16 + col) * PSTR], hi);
    const v16h bv0 = load_half_frag(vrow0 + j0, hi);
    const v16h bv1 = load_half_frag(vrow1 + j0, hi);

    acc00 = __builtin_amdgcn_wmma_f32_16x16x32_f16(false, ap0, false, bv0,
                                                   (short)0, acc00, false, false);
    acc01 = __builtin_amdgcn_wmma_f32_16x16x32_f16(false, ap0, false, bv1,
                                                   (short)0, acc01, false, false);
    acc10 = __builtin_amdgcn_wmma_f32_16x16x32_f16(false, ap1, false, bv0,
                                                   (short)0, acc10, false, false);
    acc11 = __builtin_amdgcn_wmma_f32_16x16x32_f16(false, ap1, false, bv1,
                                                   (short)0, acc11, false, false);

    // Row sums of this chunk's P (replicated across columns in C layout).
    v8f rs0 = __builtin_amdgcn_wmma_f32_16x16x32_f16(false, ap0, false, bones,
                                                     (short)0, z, false, false);
    v8f rs1 = __builtin_amdgcn_wmma_f32_16x16x32_f16(false, ap1, false, bones,
                                                     (short)0, z, false, false);
#pragma unroll
    for (int r = 0; r < 8; ++r) {
      lrow0[r] += rs0[r];
      lrow1[r] += rs1[r];
    }
    __syncthreads();
  }

#pragma unroll
  for (int r = 0; r < 8; ++r) {
    const int m = r + hi * 8;
    const float inv0 = 1.0f / lrow0[r];
    const float inv1 = 1.0f / lrow1[r];
    _Float16* orow0 =
        oH + ((size_t)(bb * SEQ + i0 + m)) * (NH * DK) + head * DK;
    _Float16* orow1 = orow0 + (size_t)16 * (NH * DK);
    orow0[col]      = (_Float16)(acc00[r] * inv0);
    orow0[16 + col] = (_Float16)(acc01[r] * inv0);
    orow1[col]      = (_Float16)(acc10[r] * inv1);
    orow1[16 + col] = (_Float16)(acc11[r] * inv1);
  }
}

// ---------------------------------------------------------------------------
// Kernel 3: output projection, transposed: D'[ch, t] = sum_k Wo[ch,k]*O[t,k]
// then out[b, ch, t] = D' + bo[ch] + x[b, ch, t] (residual), all coalesced.
// ---------------------------------------------------------------------------
__global__ __launch_bounds__(32) void out_proj_kernel(
    const _Float16* __restrict__ oH, const float* __restrict__ Wo,
    const float* __restrict__ bo, const float* __restrict__ x,
    float* __restrict__ out) {
  const int lane = threadIdx.x;
  const int hi   = lane >> 4;
  const int col  = lane & 15;
  const int m0   = blockIdx.x * 16;  // out channel tile
  const int t0   = blockIdx.y * 16;  // token tile
  const int bb   = blockIdx.z;

  const float*    wrow = Wo + (size_t)(m0 + col) * (NH * DK);          // A row
  const _Float16* orow = oH + ((size_t)(bb * SEQ + t0 + col)) * (NH * DK);

  v8f acc = {};
#pragma unroll
  for (int kc = 0; kc < NH * DK; kc += 32) {
    v16h a = load_f32_frag(wrow + kc, hi);
    v16h b = load_half_frag(orow + kc, hi);
    acc = __builtin_amdgcn_wmma_f32_16x16x32_f16(false, a, false, b,
                                                 (short)0, acc, false, false);
  }
#pragma unroll
  for (int r = 0; r < 8; ++r) {
    const int ch = m0 + r + hi * 8;
    const int t  = t0 + col;
    const size_t idx = ((size_t)(bb * CCH + ch)) * SEQ + t;
    out[idx] = acc[r] + bo[ch] + x[idx];
  }
}

// ---------------------------------------------------------------------------
extern "C" void kernel_launch(void* const* d_in, const int* in_sizes, int n_in,
                              void* d_out, int out_size, void* d_ws,
                              size_t ws_size, hipStream_t stream) {
  const float* x  = (const float*)d_in[0];
  const float* Wp = (const float*)d_in[1];
  const float* bp = (const float*)d_in[2];
  const float* Wo = (const float*)d_in[3];
  const float* bo = (const float*)d_in[4];
  float* out = (float*)d_out;

  // Workspace layout (halves): qkH 4MB | vT 2MB | oH 2MB  (8MB total)
  _Float16* qkH = (_Float16*)d_ws;
  _Float16* vT  = qkH + (size_t)BATCH * SEQ * QKW;
  _Float16* oH  = vT + (size_t)BATCH * NH * DK * SEQ;

  dim3 g1(SEQ / 16, (3 * NH * DK) / 16, BATCH);  // 128 x 24 x 4
  qkv_proj_kernel<<<g1, 32, 0, stream>>>(x, Wp, bp, qkH, vT);

  dim3 g2(SEQ / 32, NH, BATCH);                  // 64 x 4 x 4
  attn_kernel<<<g2, 32, 0, stream>>>(qkH, vT, oH);

  dim3 g3(CCH / 16, SEQ / 16, BATCH);            // 8 x 128 x 4
  out_proj_kernel<<<g3, 32, 0, stream>>>(oH, Wo, bo, x, out);
}
